// SS2D_83330955477028
// MI455X (gfx1250) — matrix operand order repaired and checked by reading, not compiled, not run on hardware
//
#include <hip/hip_runtime.h>
#include <hip/hip_bf16.h>

typedef __attribute__((ext_vector_type(16))) _Float16 v16h;
typedef __attribute__((ext_vector_type(8)))  _Float16 v8h;
typedef __attribute__((ext_vector_type(4)))  _Float16 v4h;
typedef __attribute__((ext_vector_type(8)))  float    v8f;

#define Bq   8
#define Hh   64
#define Ww   64
#define Lseq 4096
#define DM   96
#define DI   192
#define NST  16
#define Rh   6
#define DHd  32
#define Kd   4
#define Q    64
#define NC   64

// ---------- helpers ----------
__device__ __forceinline__ float silu_f(float x) { return x / (1.f + __expf(-x)); }
__device__ __forceinline__ float softplus_f(float x) {
  return (x > 20.f) ? x : log1pf(__expf(x));
}

__device__ __forceinline__ int perm_k(int k, int s) {
  switch (k) {
    case 0:  return s;
    case 1:  return ((s & (Hh - 1)) << 6) | (s >> 6);
    case 2:  return (Lseq - 1) - s;
    default: { int s1 = (Lseq - 1) - s; return ((s1 & (Hh - 1)) << 6) | (s1 >> 6); }
  }
}

__device__ __forceinline__ v4h cvt4(const float4 f) {
  v4h h;
  h[0] = (_Float16)f.x; h[1] = (_Float16)f.y;
  h[2] = (_Float16)f.z; h[3] = (_Float16)f.w;
  return h;
}

union V16U { v16h v; v8h h[2]; };

// A operand (16x32 f16): two contiguous 8-half (16B) runs per lane -> 2x ds_load_b128.
__device__ __forceinline__ v16h lds_load_a_v(const _Float16* p, int stride, int lane) {
  int hi = lane >> 4, row = lane & 15;
  const _Float16* base = p + row * stride + 8 * hi;
  V16U u;
  u.h[0] = *(const v8h*)(base);
  u.h[1] = *(const v8h*)(base + 16);
  return u.v;
}
// B operand from TRANSPOSED LDS tile Bt[n][k]: element e -> k = 8*hi + e (e<8), 16 + 8*hi + (e-8) (e>=8)
__device__ __forceinline__ v16h lds_load_bT(const _Float16* p, int stride, int lane) {
  int hi = lane >> 4, col = lane & 15;
  const _Float16* base = p + col * stride + 8 * hi;
  V16U u;
  u.h[0] = *(const v8h*)(base);
  u.h[1] = *(const v8h*)(base + 16);
  return u.v;
}

// async global->LDS 16B copy per lane (ASYNCcnt-tracked), ISA 15.18.3
__device__ __forceinline__ void async_copy_b128(void* lds_dst, const void* gsrc) {
  unsigned lds = (unsigned)(unsigned long long)lds_dst;           // LDS offset = addr[31:0]
  unsigned long long ga = (unsigned long long)gsrc;
  asm volatile("global_load_async_to_lds_b128 %0, %1, off" :: "v"(lds), "v"(ga) : "memory");
}
__device__ __forceinline__ void async_wait0() {
  asm volatile("s_wait_asynccnt 0" ::: "memory");
}

#define WMMA_F16(a, b, c) \
  __builtin_amdgcn_wmma_f32_16x16x32_f16(false, (a), false, (b), (short)0, (c), false, false)

// ---------- kernel 0: zero merge buffer ----------
__global__ void k_zero(float* __restrict__ p, int n) {
  int i = blockIdx.x * blockDim.x + threadIdx.x;
  if (i < n) p[i] = 0.f;
}

// ---------- kernel 1: in_proj GEMM (32768x96 @ 96x384), split + silu(z) ----------
__global__ __launch_bounds__(128) void k_inproj(const float* __restrict__ x,
                                                const float* __restrict__ w,
                                                float* __restrict__ xpre,
                                                float* __restrict__ z) {
  __shared__ __align__(16) _Float16 As[64][40];
  __shared__ __align__(16) _Float16 Bt[64][32];   // transposed: Bt[n][k]
  const int m0 = blockIdx.x * 64, n0 = blockIdx.y * 64;
  const int tid = threadIdx.x, lane = tid & 31, wv = tid >> 5;
  v8f acc[4] = {};
  for (int kk = 0; kk < DM; kk += 32) {
    for (int i = tid; i < 64 * 8; i += 128) {
      int r = i >> 3, c4 = (i & 7) * 4;
      *(v4h*)&As[r][c4] = cvt4(*(const float4*)&x[(size_t)(m0 + r) * DM + kk + c4]);
    }
    for (int i = tid; i < 64 * 8; i += 128) {
      int n = i >> 3, c4 = (i & 7) * 4;
      *(v4h*)&Bt[n][c4] = cvt4(*(const float4*)&w[(size_t)(n0 + n) * DM + kk + c4]);
    }
    if (kk + 32 < DM) {
      __builtin_prefetch(&x[(size_t)(m0 + (tid & 63)) * DM + kk + 32], 0, 1);
      __builtin_prefetch(&w[(size_t)(n0 + (tid & 63)) * DM + kk + 32], 0, 1);
    }
    __syncthreads();
    v16h a = lds_load_a_v(&As[wv * 16][0], 40, lane);
#pragma unroll
    for (int nt = 0; nt < 4; ++nt) {
      v16h bb = lds_load_bT(&Bt[nt * 16][0], 32, lane);
      acc[nt] = WMMA_F16(a, bb, acc[nt]);
    }
    __syncthreads();
  }
  const int hi = lane >> 4, col = lane & 15;
#pragma unroll
  for (int nt = 0; nt < 4; ++nt)
#pragma unroll
    for (int e = 0; e < 8; ++e) {
      int m = m0 + wv * 16 + e + 8 * hi;
      int n = n0 + nt * 16 + col;
      float v = acc[nt][e];
      if (n < DI) xpre[(size_t)m * DI + n] = v;
      else        z[(size_t)m * DI + (n - DI)] = silu_f(v);
    }
}

// ---------- kernel 2: depthwise 3x3 conv + bias + silu -> f16 ----------
__global__ void k_conv(const float* __restrict__ xpre, const float* __restrict__ cw,
                       const float* __restrict__ cb, _Float16* __restrict__ xh16) {
  int idx = blockIdx.x * blockDim.x + threadIdx.x;
  if (idx >= Bq * Lseq * DI) return;
  int c = idx % DI;
  int w = (idx / DI) % Ww;
  int h = (idx / (DI * Ww)) % Hh;
  int b = idx / (DI * Lseq);
  float acc = cb[c];
#pragma unroll
  for (int dy = 0; dy < 3; ++dy)
#pragma unroll
    for (int dx = 0; dx < 3; ++dx) {
      int hy = h + dy - 1, wx = w + dx - 1;
      if (hy >= 0 && hy < Hh && wx >= 0 && wx < Ww)
        acc += xpre[(size_t)(((b * Hh + hy) * Ww) + wx) * DI + c] * cw[c * 9 + dy * 3 + dx];
    }
  xh16[idx] = (_Float16)silu_f(acc);
}

// ---------- kernel 3: x_proj GEMM per (b,dir), A tile staged via async-to-LDS ----------
__global__ __launch_bounds__(128) void k_xproj(const _Float16* __restrict__ xh16,
                                               const float* __restrict__ xpw,
                                               float* __restrict__ dts,
                                               float* __restrict__ Bsb,
                                               float* __restrict__ Csb) {
  __shared__ __align__(16) _Float16 As[64][40];
  __shared__ __align__(16) _Float16 Bt[48][32];
  const int bk = blockIdx.y, b = bk >> 2, k4 = bk & 3;
  const int m0 = blockIdx.x * 64;
  const int tid = threadIdx.x, lane = tid & 31, wv = tid >> 5;
  v8f acc[3] = {};
  for (int kk = 0; kk < DI; kk += 32) {
    // A: 64 rows x 32 halfs, each row = contiguous 64B in global -> 4 async b128 per row
    for (int i = tid; i < 64 * 4; i += 128) {
      int r = i >> 2, qt = (i & 3) * 8;
      int p = perm_k(k4, m0 + r);
      async_copy_b128(&As[r][qt], &xh16[((size_t)b * Lseq + p) * DI + kk + qt]);
    }
    for (int i = tid; i < 48 * 8; i += 128) {
      int n = i >> 3, c4 = (i & 7) * 4;
      v4h hv = {};
      if (n < 38) hv = cvt4(*(const float4*)&xpw[((size_t)(k4 * 38 + n)) * DI + kk + c4]);
      *(v4h*)&Bt[n][c4] = hv;
    }
    if (kk + 32 < DI)
      __builtin_prefetch(&xh16[((size_t)b * Lseq + perm_k(k4, m0 + (tid & 63))) * DI + kk + 32], 0, 1);
    async_wait0();
    __syncthreads();
    v16h a = lds_load_a_v(&As[wv * 16][0], 40, lane);
#pragma unroll
    for (int nt = 0; nt < 3; ++nt) {
      v16h bb = lds_load_bT(&Bt[nt * 16][0], 32, lane);
      acc[nt] = WMMA_F16(a, bb, acc[nt]);
    }
    __syncthreads();
  }
  const int hi = lane >> 4, col = lane & 15;
#pragma unroll
  for (int nt = 0; nt < 3; ++nt)
#pragma unroll
    for (int e = 0; e < 8; ++e) {
      int s = m0 + wv * 16 + e + 8 * hi;
      int cc = nt * 16 + col;
      float v = acc[nt][e];
      size_t base = (size_t)bk * Lseq + s;
      if (cc < Rh)            dts[base * Rh + cc] = v;
      else if (cc < Rh + NST) Bsb[base * NST + (cc - Rh)] = v;
      else if (cc < 38)       Csb[base * NST + (cc - Rh - NST)] = v;
    }
}

// ---------- kernel 4: per-chunk stats + S_loc via WMMA (weights folded into B) ----------
__global__ __launch_bounds__(192) void k_stats(const _Float16* __restrict__ xh16,
                                               const float* __restrict__ dts,
                                               const float* __restrict__ Bsb,
                                               const float* __restrict__ Alogs,
                                               const float* __restrict__ dtb,
                                               float* __restrict__ dt_all,
                                               float* __restrict__ cum_all,
                                               float* __restrict__ ach,
                                               float* __restrict__ sloc) {
  __shared__ float s_dt[Rh][Q];
  __shared__ float s_cum[Rh][Q];
  __shared__ __align__(16) _Float16 s_Btw[Rh][NST][Q];
  __shared__ __align__(16) _Float16 s_xT[Rh][DHd][Q];
  const int bkc = blockIdx.x;
  const int cblk = bkc & (NC - 1), bk = bkc >> 6;
  const int b = bk >> 2, k4 = bk & 3;
  const int c0 = cblk * Q;
  const int tid = threadIdx.x, lane = tid & 31, wv = tid >> 5;

  // transpose-stage x: b128 global read (8 d's), scattered LDS writes
  for (int i = tid; i < Rh * Q * 4; i += 192) {
    int r = i / (Q * 4), rem = i % (Q * 4);
    int s = rem >> 2, d8 = (rem & 3) * 8;
    int p = perm_k(k4, c0 + s);
    v8h xv = *(const v8h*)&xh16[((size_t)b * Lseq + p) * DI + r * DHd + d8];
#pragma unroll
    for (int j = 0; j < 8; ++j) s_xT[r][d8 + j][s] = xv[j];
  }
  for (int i = tid; i < Rh * Q; i += 192) {
    int r = i / Q, s = i % Q;
    s_dt[r][s] = softplus_f(dts[((size_t)bk * Lseq + c0 + s) * Rh + r] + dtb[k4 * Rh + r]);
  }
  __syncthreads();
  if (tid < Rh) {
    int r = tid;
    float Ar = -__expf(Alogs[k4 * Rh + r]);
    float cacc = 0.f;
    for (int s = 0; s < Q; ++s) {
      cacc += s_dt[r][s] * Ar;
      s_cum[r][s] = cacc;
      size_t gi = ((size_t)bk * Lseq + c0 + s) * Rh + r;
      cum_all[gi] = cacc;
      dt_all[gi]  = s_dt[r][s];
    }
    ach[(size_t)(bk * NC + cblk) * Rh + r] = __expf(cacc);
  }
  __syncthreads();
  for (int i = tid; i < Rh * NST * Q; i += 192) {
    int r = i / (NST * Q), rem = i % (NST * Q);
    int n = rem >> 6, s = rem & 63;
    float wgt = __expf(s_cum[r][Q - 1] - s_cum[r][s]) * s_dt[r][s];
    s_Btw[r][n][s] = (_Float16)(wgt * Bsb[((size_t)bk * Lseq + c0 + s) * NST + n]);
  }
  __syncthreads();
  {
    const int r = wv;
    const int hi = lane >> 4, rowc = lane & 15;
#pragma unroll
    for (int mt = 0; mt < 2; ++mt) {
      v8f acc = {};
#pragma unroll
      for (int kstep = 0; kstep < 2; ++kstep) {
        int ks = kstep * 32;
        v16h a  = lds_load_a_v(&s_xT[r][mt * 16][ks], Q, lane);
        v16h bb = lds_load_bT(&s_Btw[r][0][ks], Q, lane);
        acc = WMMA_F16(a, bb, acc);
      }
      size_t base = (size_t)(bk * NC + cblk) * (Rh * DHd * NST) + (size_t)r * DHd * NST;
#pragma unroll
      for (int e = 0; e < 8; ++e) {
        int d = mt * 16 + e + 8 * hi;
        sloc[base + d * NST + rowc] = acc[e];
      }
    }
  }
}

// ---------- kernel 5: inter-chunk recurrence ----------
__global__ __launch_bounds__(256) void k_recur(const float* __restrict__ sloc,
                                               const float* __restrict__ ach,
                                               float* __restrict__ prev) {
  const int bkr = blockIdx.x;
  const int r = bkr % Rh, bk = bkr / Rh;
  const int e0 = threadIdx.x, e1 = threadIdx.x + 256;
  float st0 = 0.f, st1 = 0.f;
  for (int c = 0; c < NC; ++c) {
    size_t base = (size_t)(bk * NC + c) * (Rh * DHd * NST) + (size_t)r * DHd * NST;
    float a = ach[(size_t)(bk * NC + c) * Rh + r];
    prev[base + e0] = st0;
    prev[base + e1] = st1;
    st0 = a * st0 + sloc[base + e0];
    st1 = a * st1 + sloc[base + e1];
  }
}

// ---------- kernel 6: per-chunk output (CB, y_intra, y_inter, D*x) + merge ----------
__global__ __launch_bounds__(256) void k_chunk(const _Float16* __restrict__ xh16,
                                               const float* __restrict__ Bsb,
                                               const float* __restrict__ Csb,
                                               const float* __restrict__ cum_all,
                                               const float* __restrict__ dt_all,
                                               const float* __restrict__ prev,
                                               const float* __restrict__ Ds,
                                               float* __restrict__ ymrg) {
  __shared__ __align__(16) _Float16 s_C[Q][32];
  __shared__ __align__(16) _Float16 s_Bc[Q][32];
  __shared__ __align__(16) _Float16 s_CB[Q][Q];
  __shared__ __align__(16) _Float16 s_xT[Rh][DHd][Q];
  __shared__ __align__(16) _Float16 s_p[Rh][DHd][32];
  __shared__ float s_cum[Rh][Q];
  __shared__ float s_dt[Rh][Q];
  const int bkc = blockIdx.x;
  const int cblk = bkc & (NC - 1), bk = bkc >> 6;
  const int b = bk >> 2, k4 = bk & 3;
  const int c0 = cblk * Q;
  const int tid = threadIdx.x, lane = tid & 31, wv = tid >> 5;
  const int hi = lane >> 4, colc = lane & 15;

  for (int i = tid; i < Q * 8; i += 256) {
    int t = i >> 3, n4 = (i & 7) * 4;
    v4h cv = {}, bv = {};
    if (n4 < NST) {
      cv = cvt4(*(const float4*)&Csb[((size_t)bk * Lseq + c0 + t) * NST + n4]);
      bv = cvt4(*(const float4*)&Bsb[((size_t)bk * Lseq + c0 + t) * NST + n4]);
    }
    *(v4h*)&s_C[t][n4]  = cv;
    *(v4h*)&s_Bc[t][n4] = bv;
  }
  for (int i = tid; i < Rh * Q * 4; i += 256) {
    int r = i / (Q * 4), rem = i % (Q * 4);
    int s = rem >> 2, d8 = (rem & 3) * 8;
    int p = perm_k(k4, c0 + s);
    v8h xv = *(const v8h*)&xh16[((size_t)b * Lseq + p) * DI + r * DHd + d8];
#pragma unroll
    for (int j = 0; j < 8; ++j) s_xT[r][d8 + j][s] = xv[j];
  }
  for (int i = tid; i < Rh * DHd * 8; i += 256) {
    int r = i / (DHd * 8), rem = i % (DHd * 8);
    int d = rem >> 3, n4 = (rem & 7) * 4;
    v4h pv = {};
    if (n4 < NST) {
      size_t base = (size_t)(bk * NC + cblk) * (Rh * DHd * NST) + (size_t)r * DHd * NST;
      pv = cvt4(*(const float4*)&prev[base + d * NST + n4]);
    }
    *(v4h*)&s_p[r][d][n4] = pv;
  }
  for (int i = tid; i < Rh * Q; i += 256) {
    int r = i / Q, s = i % Q;
    size_t gi = ((size_t)bk * Lseq + c0 + s) * Rh + r;
    s_cum[r][s] = cum_all[gi];
    s_dt[r][s]  = dt_all[gi];
  }
  __syncthreads();

  // CB = C (64x16) . B^T (16x64), K padded to 32
  for (int t8 = wv; t8 < 16; t8 += 8) {
    int mt = t8 >> 2, nt = t8 & 3;
    v16h a  = lds_load_a_v(&s_C[mt * 16][0], 32, lane);
    v16h bb = lds_load_bT(&s_Bc[nt * 16][0], 32, lane);
    v8f cbacc = {};
    cbacc = WMMA_F16(a, bb, cbacc);
#pragma unroll
    for (int e = 0; e < 8; ++e)
      s_CB[mt * 16 + e + 8 * hi][nt * 16 + colc] = (_Float16)cbacc[e];
  }
  __syncthreads();

  if (wv < Rh) {
    const int r = wv;
#pragma unroll
    for (int mt = 0; mt < 4; ++mt) {
      const int trow = mt * 16 + colc;
      const float cum_t = s_cum[r][trow];
#pragma unroll
      for (int nt = 0; nt < 2; ++nt) {
        v8f acc = {};
#pragma unroll
        for (int kstep = 0; kstep < 2; ++kstep) {
          int ks = kstep * 32;
          const _Float16* cbrow = &s_CB[trow][ks + 8 * hi];
          v8h cb0 = *(const v8h*)(cbrow);
          v8h cb1 = *(const v8h*)(cbrow + 16);
          v16h a;
#pragma unroll
          for (int e = 0; e < 16; ++e) {
            int k = (e < 8) ? (e + 8 * hi) : (e + 8 + 8 * hi);
            int s = ks + k;
            float cbv = (e < 8) ? (float)cb0[e & 7] : (float)cb1[e & 7];
            float v = 0.f;
            if (s <= trow)
              v = cbv * __expf(cum_t - s_cum[r][s]) * s_dt[r][s];
            a[e] = (_Float16)v;
          }
          v16h bb = lds_load_bT(&s_xT[r][nt * 16][ks], Q, lane);
          acc = WMMA_F16(a, bb, acc);
        }
        {
          v16h a2 = lds_load_a_v(&s_C[mt * 16][0], 32, lane);
          a2 = a2 * (_Float16)__expf(cum_t);
          v16h b2 = lds_load_bT(&s_p[r][nt * 16][0], 32, lane);
          acc = WMMA_F16(a2, b2, acc);
        }
#pragma unroll
        for (int e = 0; e < 8; ++e) {
          int tl = mt * 16 + e + 8 * hi;
          int d  = nt * 16 + colc;
          int sg = c0 + tl;
          int p  = perm_k(k4, sg);
          float val = acc[e] + Ds[(k4 * Rh + r) * DHd + d] * (float)s_xT[r][d][tl];
          atomicAdd(&ymrg[((size_t)b * Lseq + p) * DI + r * DHd + d], val);
        }
      }
    }
  }
}

// ---------- kernel 7: layernorm * z -> f16 ----------
__global__ __launch_bounds__(192) void k_ln(const float* __restrict__ ymrg,
                                            const float* __restrict__ z,
                                            const float* __restrict__ lw,
                                            const float* __restrict__ lb,
                                            _Float16* __restrict__ yln) {
  __shared__ float red[DI];
  const int m = blockIdx.x, t = threadIdx.x;
  float v = ymrg[(size_t)m * DI + t];
  red[t] = v;
  __syncthreads();
  if (t < 64) red[t] = red[t] + red[t + 64] + red[t + 128];
  __syncthreads();
  for (int off = 32; off > 0; off >>= 1) {
    if (t < off) red[t] += red[t + off];
    __syncthreads();
  }
  float mu = red[0] / (float)DI;
  __syncthreads();
  float dv = v - mu;
  red[t] = dv * dv;
  __syncthreads();
  if (t < 64) red[t] = red[t] + red[t + 64] + red[t + 128];
  __syncthreads();
  for (int off = 32; off > 0; off >>= 1) {
    if (t < off) red[t] += red[t + off];
    __syncthreads();
  }
  float var = red[0] / (float)DI;
  float ln = dv * rsqrtf(var + 1e-5f) * lw[t] + lb[t];
  yln[(size_t)m * DI + t] = (_Float16)(ln * z[(size_t)m * DI + t]);
}

// ---------- kernel 8: out_proj GEMM (32768x192 @ 192x96), A staged async ----------
__global__ __launch_bounds__(128) void k_outproj(const _Float16* __restrict__ yln,
                                                 const float* __restrict__ w,
                                                 float* __restrict__ out) {
  __shared__ __align__(16) _Float16 As[64][40];
  __shared__ __align__(16) _Float16 Bt[48][32];
  const int m0 = blockIdx.x * 64, n0 = blockIdx.y * 48;
  const int tid = threadIdx.x, lane = tid & 31, wv = tid >> 5;
  v8f acc[3] = {};
  for (int kk = 0; kk < DI; kk += 32) {
    for (int i = tid; i < 64 * 4; i += 128) {
      int r = i >> 2, qt = (i & 3) * 8;
      async_copy_b128(&As[r][qt], &yln[(size_t)(m0 + r) * DI + kk + qt]);
    }
    for (int i = tid; i < 48 * 8; i += 128) {
      int n = i >> 3, c4 = (i & 7) * 4;
      *(v4h*)&Bt[n][c4] = cvt4(*(const float4*)&w[(size_t)(n0 + n) * DI + kk + c4]);
    }
    if (kk + 32 < DI)
      __builtin_prefetch(&yln[(size_t)(m0 + (tid & 63)) * DI + kk + 32], 0, 1);
    async_wait0();
    __syncthreads();
    v16h a = lds_load_a_v(&As[wv * 16][0], 40, lane);
#pragma unroll
    for (int nt = 0; nt < 3; ++nt) {
      v16h bb = lds_load_bT(&Bt[nt * 16][0], 32, lane);
      acc[nt] = WMMA_F16(a, bb, acc[nt]);
    }
    __syncthreads();
  }
  const int hi = lane >> 4, col = lane & 15;
#pragma unroll
  for (int nt = 0; nt < 3; ++nt)
#pragma unroll
    for (int e = 0; e < 8; ++e) {
      int m = m0 + wv * 16 + e + 8 * hi;
      int n = n0 + nt * 16 + col;
      out[(size_t)m * DM + n] = acc[nt][e];
    }
}

// ---------- launch ----------
extern "C" void kernel_launch(void* const* d_in, const int* in_sizes, int n_in,
                              void* d_out, int out_size, void* d_ws, size_t ws_size,
                              hipStream_t stream) {
  const float* x    = (const float*)d_in[0];
  const float* ipw  = (const float*)d_in[1];
  const float* cw   = (const float*)d_in[2];
  const float* cb   = (const float*)d_in[3];
  const float* xpw  = (const float*)d_in[4];
  const float* alog = (const float*)d_in[5];
  const float* dsp  = (const float*)d_in[6];
  const float* dtb  = (const float*)d_in[7];
  const float* lw   = (const float*)d_in[8];
  const float* lb   = (const float*)d_in[9];
  const float* opw  = (const float*)d_in[10];
  float* out = (float*)d_out;

  char* ws = (char*)d_ws;
  size_t off = 0;
  auto alloc = [&](size_t bytes) { char* p = ws + off; off += (bytes + 255) & ~(size_t)255; return p; };
  float*    XPRE = (float*)   alloc((size_t)Bq * Lseq * DI * 4);
  float*    Z    = (float*)   alloc((size_t)Bq * Lseq * DI * 4);
  _Float16* XH16 = (_Float16*)alloc((size_t)Bq * Lseq * DI * 2);
  float*    DTS  = (float*)   alloc((size_t)Bq * Kd * Lseq * Rh * 4);
  float*    DT   = (float*)   alloc((size_t)Bq * Kd * Lseq * Rh * 4);
  float*    CUM  = (float*)   alloc((size_t)Bq * Kd * Lseq * Rh * 4);
  float*    BSB  = (float*)   alloc((size_t)Bq * Kd * Lseq * NST * 4);
  float*    CSB  = (float*)   alloc((size_t)Bq * Kd * Lseq * NST * 4);
  float*    ACH  = (float*)   alloc((size_t)Bq * Kd * NC * Rh * 4);
  float*    SLOC = (float*)   alloc((size_t)Bq * Kd * NC * Rh * DHd * NST * 4);
  float*    PREV = (float*)   alloc((size_t)Bq * Kd * NC * Rh * DHd * NST * 4);
  float*    YMRG = (float*)   alloc((size_t)Bq * Lseq * DI * 4);
  _Float16* YLN  = (_Float16*)alloc((size_t)Bq * Lseq * DI * 2);

  const int nElem = Bq * Lseq * DI;
  k_zero<<<(nElem + 255) / 256, 256, 0, stream>>>(YMRG, nElem);
  k_inproj<<<dim3((Bq * Lseq) / 64, (2 * DI) / 64), 128, 0, stream>>>(x, ipw, XPRE, Z);
  k_conv<<<(nElem + 255) / 256, 256, 0, stream>>>(XPRE, cw, cb, XH16);
  k_xproj<<<dim3(Lseq / 64, Bq * Kd), 128, 0, stream>>>(XH16, xpw, DTS, BSB, CSB);
  k_stats<<<Bq * Kd * NC, 192, 0, stream>>>(XH16, DTS, BSB, alog, dtb, DT, CUM, ACH, SLOC);
  k_recur<<<Bq * Kd * Rh, 256, 0, stream>>>(SLOC, ACH, PREV);
  k_chunk<<<Bq * Kd * NC, 256, 0, stream>>>(XH16, BSB, CSB, CUM, DT, PREV, dsp, YMRG);
  k_ln<<<Bq * Lseq, DI, 0, stream>>>(YMRG, Z, lw, lb, YLN);
  k_outproj<<<dim3((Bq * Lseq) / 64, 2), 128, 0, stream>>>(YLN, opw, out);
}
